// PerformerCrossAttention_29618094473989
// MI455X (gfx1250) — matrix operand order repaired and checked by reading, not compiled, not run on hardware
//
#include <hip/hip_runtime.h>
#include <hip/hip_bf16.h>

// ---------------------------------------------------------------------------
// Performer cross-attention for gfx1250 (MI455X), bf16 WMMA pipeline.
// D = A x B + C with v_wmma_f32_16x16x32_bf16 (fp32 accumulate).
// Global->LDS staging uses GLOBAL_LOAD_ASYNC_TO_LDS_B128 when the builtin
// exists (ASYNCcnt path), else register staging.
// ---------------------------------------------------------------------------

typedef __attribute__((ext_vector_type(16))) __bf16 v16bf;
typedef __attribute__((ext_vector_type(8)))  __bf16 v8bf;
typedef __attribute__((ext_vector_type(8)))  float  v8f;
typedef __attribute__((ext_vector_type(4)))  int    v4i;

#define D_MODEL 1024
#define SEQ     4096
#define NHEAD   16
#define D_HEAD  64
#define M_TOT   16384   // B * SEQ
#define EPS_F   1e-6f

#if __has_builtin(__builtin_amdgcn_global_load_async_to_lds_b128) && \
    __has_builtin(__builtin_amdgcn_s_wait_asynccnt)
#define HAVE_ASYNC_LDS 1
#else
#define HAVE_ASYNC_LDS 0
#endif

__device__ __forceinline__ void async_copy16(void* lds, const void* g) {
#if HAVE_ASYNC_LDS
    __builtin_amdgcn_global_load_async_to_lds_b128(
        (__attribute__((address_space(1))) v4i*)(void*)(size_t)g,
        (__attribute__((address_space(3))) v4i*)lds, 0, 0);
#else
    (void)lds; (void)g;
#endif
}

__device__ __forceinline__ void async_wait0() {
#if HAVE_ASYNC_LDS
    __builtin_amdgcn_s_wait_asynccnt(0);
#endif
}

__device__ __forceinline__ v8f wmma_bf16(v16bf a, v16bf b, v8f c) {
    return __builtin_amdgcn_wmma_f32_16x16x32_bf16(
        false, a, false, b, (short)0, c, false, false);
}

// elu(x)+1 :  x>0 ? x+1 : exp(x)
__device__ __forceinline__ float feat_map(float x) {
    return x > 0.f ? x + 1.f : __expf(x);
}

// Build a 16-element bf16 fragment from two 16-byte chunks.
__device__ __forceinline__ v16bf frag_from(const __bf16* lo_p, const __bf16* hi_p) {
    v8bf lo = *(const v8bf*)lo_p;
    v8bf hh = *(const v8bf*)hi_p;
    v16bf r;
#pragma unroll
    for (int i = 0; i < 8; ++i) { r[i] = lo[i]; r[i + 8] = hh[i]; }
    return r;
}

// ---------------------------------------------------------------------------
// fp32 -> bf16 bulk convert (vectorized x4)
// ---------------------------------------------------------------------------
__global__ void cvt_f32_bf16(const float* __restrict__ src,
                             __bf16* __restrict__ dst, int n) {
    int i = (blockIdx.x * blockDim.x + threadIdx.x) * 4;
    if (i + 3 < n) {
        float4 f = *(const float4*)(src + i);
        __bf16 p[4] __attribute__((aligned(8)));
        p[0] = (__bf16)f.x; p[1] = (__bf16)f.y;
        p[2] = (__bf16)f.z; p[3] = (__bf16)f.w;
        *(int2*)(dst + i) = *(const int2*)p;
    } else {
        for (int j = i; j < n; ++j) dst[j] = (__bf16)src[j];
    }
}

// ---------------------------------------------------------------------------
// Generic projection GEMM:  out = X[M,1024] * W[1024,1024]^T + bias
//   MODE 0: Q  -> feat_map, store bf16 natural [M,1024]
//   MODE 1: K  -> mask, feat_map, store bf16 transposed [bh][d][SEQ]
//   MODE 2: V  -> mask,           store bf16 transposed [bh][d][SEQ]
//   MODE 3: O  -> store fp32 natural [M,1024] (final output)
// Block 128x128, K-step 32, 8 waves (2x4 16x16 tiles each), double-buffered
// LDS (pitch 40 bf16 = 80B, conflict-free b128 reads), one barrier per step.
// ---------------------------------------------------------------------------
#define BLK   128
#define BK    32
#define LDSP  40
#define NKT   (D_MODEL / BK)   // 32

template <int MODE>
__global__ __launch_bounds__(256)
void gemm_kernel(const __bf16* __restrict__ X,
                 const __bf16* __restrict__ W,
                 const float*  __restrict__ bias,
                 const unsigned char* __restrict__ mask,  // [M] (modes 1,2)
                 __bf16* __restrict__ outB,
                 float*  __restrict__ outF) {
    __shared__ __bf16 As[2][BLK][LDSP] __attribute__((aligned(16)));
    __shared__ __bf16 Bs[2][BLK][LDSP] __attribute__((aligned(16)));

    const int t    = threadIdx.x;
    const int lane = t & 31;
    const int wave = t >> 5;
    const int l16  = lane & 15;
    const int hi   = lane >> 4;          // 0/1
    const int wm   = wave & 3;           // M group (32 rows)
    const int wn   = wave >> 2;          // N group (64 cols)

    const int m0 = blockIdx.x * BLK;
    const int n0 = blockIdx.y * BLK;

    // staging: each thread moves 16 bf16 (32B) of A and of B per K-step
    const int lr = t >> 1;
    const int lc = (t & 1) * 16;
    const __bf16* Ag = X + (size_t)(m0 + lr) * D_MODEL + lc;
    const __bf16* Bg = W + (size_t)(n0 + lr) * D_MODEL + lc;

    v8f acc[2][4] = {};
    int buf = 0;

#if HAVE_ASYNC_LDS
    async_copy16(&As[0][lr][lc],     Ag);
    async_copy16(&As[0][lr][lc + 8], Ag + 8);
    async_copy16(&Bs[0][lr][lc],     Bg);
    async_copy16(&Bs[0][lr][lc + 8], Bg + 8);
    async_wait0();
    __syncthreads();
#else
    int4 ra0, ra1, rb0, rb1;
    ra0 = *(const int4*)(Ag);     ra1 = *(const int4*)(Ag + 8);
    rb0 = *(const int4*)(Bg);     rb1 = *(const int4*)(Bg + 8);
    *(int4*)&As[0][lr][lc] = ra0;  *(int4*)&As[0][lr][lc + 8] = ra1;
    *(int4*)&Bs[0][lr][lc] = rb0;  *(int4*)&Bs[0][lr][lc + 8] = rb1;
    __syncthreads();
#endif

    for (int kt = 0; kt < NKT; ++kt) {
        const bool has_next = (kt + 1 < NKT);
        if (has_next) {
            const __bf16* a = Ag + (kt + 1) * BK;
            const __bf16* b = Bg + (kt + 1) * BK;
#if HAVE_ASYNC_LDS
            const int nb = buf ^ 1;
            async_copy16(&As[nb][lr][lc],     a);
            async_copy16(&As[nb][lr][lc + 8], a + 8);
            async_copy16(&Bs[nb][lr][lc],     b);
            async_copy16(&Bs[nb][lr][lc + 8], b + 8);
#else
            ra0 = *(const int4*)(a);      ra1 = *(const int4*)(a + 8);
            rb0 = *(const int4*)(b);      rb1 = *(const int4*)(b + 8);
#endif
            if (kt + 2 < NKT) {            // gfx1250 global_prefetch_b8 path
                __builtin_prefetch(Ag + (kt + 2) * BK, 0, 0);
                __builtin_prefetch(Bg + (kt + 2) * BK, 0, 0);
            }
        }

        v16bf af[2], bfrag[4];
#pragma unroll
        for (int mt = 0; mt < 2; ++mt) {
            const int row = wm * 32 + mt * 16 + l16;
            af[mt] = frag_from(&As[buf][row][hi * 8], &As[buf][row][16 + hi * 8]);
        }
#pragma unroll
        for (int nt = 0; nt < 4; ++nt) {
            const int row = wn * 64 + nt * 16 + l16;
            bfrag[nt] = frag_from(&Bs[buf][row][hi * 8], &Bs[buf][row][16 + hi * 8]);
        }
#pragma unroll
        for (int mt = 0; mt < 2; ++mt)
#pragma unroll
            for (int nt = 0; nt < 4; ++nt)
                acc[mt][nt] = wmma_bf16(af[mt], bfrag[nt], acc[mt][nt]);

        if (has_next) {
#if HAVE_ASYNC_LDS
            async_wait0();
#else
            const int nb = buf ^ 1;
            *(int4*)&As[nb][lr][lc] = ra0;  *(int4*)&As[nb][lr][lc + 8] = ra1;
            *(int4*)&Bs[nb][lr][lc] = rb0;  *(int4*)&Bs[nb][lr][lc + 8] = rb1;
#endif
        }
        __syncthreads();
        buf ^= 1;
    }

    // ----- epilogue -----
#pragma unroll
    for (int mt = 0; mt < 2; ++mt) {
#pragma unroll
        for (int nt = 0; nt < 4; ++nt) {
            const int nG    = n0 + wn * 64 + nt * 16 + l16;
            const float bs  = bias[nG];
            const int mTile = m0 + wm * 32 + mt * 16;   // never straddles a batch

            if (MODE == 0) {                 // Q: feat map, natural bf16
#pragma unroll
                for (int r = 0; r < 8; ++r) {
                    const int mG = mTile + r + 8 * hi;
                    outB[(size_t)mG * D_MODEL + nG] =
                        (__bf16)feat_map(acc[mt][nt][r] + bs);
                }
            } else if (MODE == 3) {          // O: fp32 natural
#pragma unroll
                for (int r = 0; r < 8; ++r) {
                    const int mG = mTile + r + 8 * hi;
                    outF[(size_t)mG * D_MODEL + nG] = acc[mt][nt][r] + bs;
                }
            } else {                         // K/V: mask (+feat), transposed
                const int b  = mTile >> 12;      // SEQ tokens per batch
                const int h  = nG >> 6;
                const int d  = nG & 63;
                const int l0 = (mTile & (SEQ - 1)) + 8 * hi;
                __bf16 pack[8] __attribute__((aligned(16)));
#pragma unroll
                for (int r = 0; r < 8; ++r) {
                    const int mG = mTile + r + 8 * hi;
                    float vv = acc[mt][nt][r] + bs;
                    vv *= (mask[mG] ? 0.f : 1.f);    // True = padded
                    if (MODE == 1) vv = feat_map(vv);
                    pack[r] = (__bf16)vv;
                }
                __bf16* dst = outB +
                    ((size_t)((b * NHEAD + h) * D_HEAD + d) * SEQ) + l0;
                *(int4*)dst = *(const int4*)pack;
            }
        }
    }
}

// ---------------------------------------------------------------------------
// k_sum[bh*64+d] = sum_l kfT[bh][d][l]   (one wave per row)
// ---------------------------------------------------------------------------
__global__ __launch_bounds__(256)
void ksum_kernel(const __bf16* __restrict__ kfT, float* __restrict__ ksum) {
    const int row  = blockIdx.x * 8 + (threadIdx.x >> 5);   // 0..4095
    const int lane = threadIdx.x & 31;
    const __bf16* p = kfT + (size_t)row * SEQ;
    float s = 0.f;
    for (int c = lane; c < SEQ / 8; c += 32) {
        v8bf v = *(const v8bf*)(p + c * 8);
#pragma unroll
        for (int i = 0; i < 8; ++i) s += (float)v[i];
    }
#pragma unroll
    for (int off = 16; off > 0; off >>= 1) s += __shfl_xor(s, off, 32);
    if (lane == 0) ksum[row] = s;
}

// ---------------------------------------------------------------------------
// kv partials: kf^T @ vh over a K slice of SEQ/4. 256 blocks (4 slices x 64
// bh), 4 waves each (one 16-col N tile x 4 M tiles). Software-pipelined 2x:
// next-step fragments load while current WMMAs execute. fp32 partial store
// TRANSPOSED: kvPart[slice][bh][n][d].
// ---------------------------------------------------------------------------
#define KV_SPLIT 4
#define KV_CHUNK (SEQ / KV_SPLIT)   // 1024

__global__ __launch_bounds__(128)
void kv_kernel(const __bf16* __restrict__ kfT,
               const __bf16* __restrict__ vhT,
               float* __restrict__ kvPart) {
    const int bh   = blockIdx.x & 63;
    const int sl   = blockIdx.x >> 6;
    const int lane = threadIdx.x & 31;
    const int w    = threadIdx.x >> 5;     // N tile
    const int l16  = lane & 15;
    const int hi   = lane >> 4;
    const int k0   = sl * KV_CHUNK;

    const __bf16* A  = kfT + (size_t)bh * D_HEAD * SEQ;
    const __bf16* Bv = vhT + (size_t)bh * D_HEAD * SEQ + (size_t)(w * 16 + l16) * SEQ;

    const __bf16* aRow[4];
#pragma unroll
    for (int mt = 0; mt < 4; ++mt)
        aRow[mt] = A + (size_t)(mt * 16 + l16) * SEQ;

    v8f acc[4] = {};
    v16bf a0[4], a1[4], b0, b1;

    // prologue: fragments for k0
    b0 = frag_from(Bv + k0 + hi * 8, Bv + k0 + hi * 8 + 16);
#pragma unroll
    for (int mt = 0; mt < 4; ++mt)
        a0[mt] = frag_from(aRow[mt] + k0 + hi * 8, aRow[mt] + k0 + hi * 8 + 16);

    for (int k = k0; k < k0 + KV_CHUNK; k += 64) {
        // stage k+32 while computing with k
        b1 = frag_from(Bv + k + 32 + hi * 8, Bv + k + 32 + hi * 8 + 16);
#pragma unroll
        for (int mt = 0; mt < 4; ++mt)
            a1[mt] = frag_from(aRow[mt] + k + 32 + hi * 8,
                               aRow[mt] + k + 32 + hi * 8 + 16);
#pragma unroll
        for (int mt = 0; mt < 4; ++mt)
            acc[mt] = wmma_bf16(a0[mt], b0, acc[mt]);

        if (k + 64 < k0 + KV_CHUNK) {
            b0 = frag_from(Bv + k + 64 + hi * 8, Bv + k + 64 + hi * 8 + 16);
#pragma unroll
            for (int mt = 0; mt < 4; ++mt)
                a0[mt] = frag_from(aRow[mt] + k + 64 + hi * 8,
                                   aRow[mt] + k + 64 + hi * 8 + 16);
            __builtin_prefetch(Bv + k + 96, 0, 0);
        }
#pragma unroll
        for (int mt = 0; mt < 4; ++mt)
            acc[mt] = wmma_bf16(a1[mt], b1, acc[mt]);
    }

    // store fp32 partials, transposed: kvPart[sl][bh][n][d]
    float* base = kvPart + (((size_t)sl * 64 + bh) * D_HEAD +
                            (size_t)(w * 16 + l16)) * D_HEAD;
#pragma unroll
    for (int mt = 0; mt < 4; ++mt) {
        float pk[8] __attribute__((aligned(16)));
#pragma unroll
        for (int r = 0; r < 8; ++r) pk[r] = acc[mt][r];
        float* dst = base + mt * 16 + 8 * hi;
        *(float4*)dst       = *(const float4*)pk;
        *(float4*)(dst + 4) = *(const float4*)(pk + 4);
    }
}

// reduce 4 fp32 partial slices -> bf16 kvT[bh][n][d]
__global__ __launch_bounds__(256)
void kv_reduce_kernel(const float* __restrict__ kvPart,
                      __bf16* __restrict__ kvT) {
    const int n1 = 64 * D_HEAD * D_HEAD;                 // 262144
    int i = (blockIdx.x * blockDim.x + threadIdx.x) * 4;
    if (i + 3 < n1) {
        float4 s = *(const float4*)(kvPart + i);
#pragma unroll
        for (int sl = 1; sl < KV_SPLIT; ++sl) {
            float4 p = *(const float4*)(kvPart + (size_t)sl * n1 + i);
            s.x += p.x; s.y += p.y; s.z += p.z; s.w += p.w;
        }
        __bf16 pk[4] __attribute__((aligned(8)));
        pk[0] = (__bf16)s.x; pk[1] = (__bf16)s.y;
        pk[2] = (__bf16)s.z; pk[3] = (__bf16)s.w;
        *(int2*)(kvT + i) = *(const int2*)pk;
    }
}

// ---------------------------------------------------------------------------
// out = (qf @ kv) / (qf . k_sum + eps), merged-head bf16 [M,1024].
// Block: 128 tokens x 1 head; 8 waves = 8 M tiles, 4 N tiles each, K = 64.
// ---------------------------------------------------------------------------
__global__ __launch_bounds__(256)
void out_kernel(const __bf16* __restrict__ qf,     // [M,1024]
                const __bf16* __restrict__ kvT,    // [bh][64][64]
                const float*  __restrict__ ksum,   // [bh][64]
                __bf16* __restrict__ merged) {     // [M,1024]
    __shared__ float den[128];
    const int tok0 = blockIdx.x * 128;
    const int h    = blockIdx.y;
    const int b    = tok0 >> 12;
    const int bh   = b * NHEAD + h;
    const int t    = threadIdx.x;

    if (t < 128) {                                 // denominators
        const __bf16* qp = qf + (size_t)(tok0 + t) * D_MODEL + h * D_HEAD;
        const float*  kp = ksum + bh * D_HEAD;
        float s = EPS_F;
#pragma unroll
        for (int c = 0; c < 8; ++c) {
            v8bf qv = *(const v8bf*)(qp + c * 8);
#pragma unroll
            for (int i = 0; i < 8; ++i) s += (float)qv[i] * kp[c * 8 + i];
        }
        den[t] = s;
    }
    __syncthreads();

    const int lane = t & 31, wave = t >> 5;
    const int l16 = lane & 15, hi = lane >> 4;
    const __bf16* kvb = kvT + (size_t)bh * D_HEAD * D_HEAD;

    v8f acc[4] = {};
#pragma unroll
    for (int ks = 0; ks < D_HEAD; ks += BK) {
        const __bf16* ab = qf + (size_t)(tok0 + wave * 16 + l16) * D_MODEL +
                           h * D_HEAD + ks + hi * 8;
        v16bf afrag = frag_from(ab, ab + 16);
#pragma unroll
        for (int nt = 0; nt < 4; ++nt) {
            const __bf16* bb = kvb + (size_t)(nt * 16 + l16) * D_HEAD + ks + hi * 8;
            v16bf bfrag = frag_from(bb, bb + 16);
            acc[nt] = wmma_bf16(afrag, bfrag, acc[nt]);
        }
    }
#pragma unroll
    for (int nt = 0; nt < 4; ++nt) {
        const int col = h * D_HEAD + nt * 16 + l16;
#pragma unroll
        for (int r = 0; r < 8; ++r) {
            const int tl = wave * 16 + r + 8 * hi;
            merged[(size_t)(tok0 + tl) * D_MODEL + col] =
                (__bf16)(acc[nt][r] / den[tl]);
        }
    }
}

// ---------------------------------------------------------------------------
extern "C" void kernel_launch(void* const* d_in, const int* in_sizes, int n_in,
                              void* d_out, int out_size, void* d_ws, size_t ws_size,
                              hipStream_t stream) {
    (void)in_sizes; (void)n_in; (void)out_size; (void)ws_size;

    const float* q    = (const float*)d_in[0];
    const float* k    = (const float*)d_in[1];
    const float* v    = (const float*)d_in[2];
    const unsigned char* mask = (const unsigned char*)d_in[3];  // bool [B,Lk]
    const float* Wq   = (const float*)d_in[4];
    const float* bq   = (const float*)d_in[5];
    const float* Wk   = (const float*)d_in[6];
    const float* bk   = (const float*)d_in[7];
    const float* Wv   = (const float*)d_in[8];
    const float* bv   = (const float*)d_in[9];
    const float* Wo   = (const float*)d_in[10];
    const float* bo   = (const float*)d_in[11];
    float* out = (float*)d_out;

    char* ws = (char*)d_ws;
    const size_t MB = 1024ull * 1024ull;
    __bf16* wWq    = (__bf16*)(ws + 0 * MB);      // 2 MB each
    __bf16* wWk    = (__bf16*)(ws + 2 * MB);
    __bf16* wWv    = (__bf16*)(ws + 4 * MB);
    __bf16* wWo    = (__bf16*)(ws + 6 * MB);
    __bf16* Xbf    = (__bf16*)(ws + 8 * MB);      // 32 MB (reused q->k->v)
    __bf16* qf     = (__bf16*)(ws + 40 * MB);     // 32 MB
    __bf16* kfT    = (__bf16*)(ws + 72 * MB);     // 32 MB
    __bf16* vhT    = (__bf16*)(ws + 104 * MB);    // 32 MB
    __bf16* merged = (__bf16*)(ws + 136 * MB);    // 32 MB
    __bf16* kvT    = (__bf16*)(ws + 168 * MB);    // 512 KB
    float*  ksum   = (float*)(ws + 169 * MB);     // 16 KB
    float*  kvPart = (float*)(ws + 170 * MB);     // 4 MB

    const int NW = D_MODEL * D_MODEL;      // 1 M elems
    const int NX = M_TOT * D_MODEL;        // 16 M elems
    const int CT = 256;
    const int gW = (NW / 4 + CT - 1) / CT;
    const int gX = (NX / 4 + CT - 1) / CT;

    cvt_f32_bf16<<<gW, CT, 0, stream>>>(Wq, wWq, NW);
    cvt_f32_bf16<<<gW, CT, 0, stream>>>(Wk, wWk, NW);
    cvt_f32_bf16<<<gW, CT, 0, stream>>>(Wv, wWv, NW);
    cvt_f32_bf16<<<gW, CT, 0, stream>>>(Wo, wWo, NW);

    dim3 gg(M_TOT / BLK, D_MODEL / BLK);   // 128 x 8

    cvt_f32_bf16<<<gX, CT, 0, stream>>>(q, Xbf, NX);
    gemm_kernel<0><<<gg, 256, 0, stream>>>(Xbf, wWq, bq, nullptr, qf, nullptr);

    cvt_f32_bf16<<<gX, CT, 0, stream>>>(k, Xbf, NX);
    gemm_kernel<1><<<gg, 256, 0, stream>>>(Xbf, wWk, bk, mask, kfT, nullptr);

    cvt_f32_bf16<<<gX, CT, 0, stream>>>(v, Xbf, NX);
    gemm_kernel<2><<<gg, 256, 0, stream>>>(Xbf, wWv, bv, mask, vhT, nullptr);

    ksum_kernel<<<512, 256, 0, stream>>>(kfT, ksum);
    kv_kernel<<<KV_SPLIT * 64, 128, 0, stream>>>(kfT, vhT, kvPart);
    kv_reduce_kernel<<<(64 * D_HEAD * D_HEAD / 4 + 255) / 256, 256, 0, stream>>>(kvPart, kvT);
    out_kernel<<<dim3(M_TOT / 128, NHEAD), 256, 0, stream>>>(qf, kvT, ksum, merged);

    gemm_kernel<3><<<gg, 256, 0, stream>>>(merged, wWo, bo, nullptr, nullptr, out);
}
